// TransformerBlock_17772574671213
// MI455X (gfx1250) — matrix operand order repaired and checked by reading, not compiled, not run on hardware
//
#include <hip/hip_runtime.h>

// ---------------- constants (match reference) ----------------
#define Bb   2
#define Ss   2048
#define Dd   2048
#define Hh   16
#define HKk  4
#define HDd  128
#define HIDd 8192
#define WINw 1024
#define SINKs 4

typedef __attribute__((ext_vector_type(16))) __bf16 v16bf;
typedef __attribute__((ext_vector_type(8)))  float  v8f;

union U16 {
    v16bf v;
    uint4 q[2];
    unsigned short us[16];
};

__device__ __forceinline__ unsigned short f2bf(float f) {
    unsigned u = __float_as_uint(f);
    unsigned r = u + 0x7FFFu + ((u >> 16) & 1u);
    return (unsigned short)(r >> 16);
}
__device__ __forceinline__ v8f zero8() {
    v8f z;
#pragma unroll
    for (int i = 0; i < 8; ++i) z[i] = 0.0f;
    return z;
}

// ---------------- tiled transpose + fp32 -> bf16 cast ----------------------
// src: [K,N] fp32 row-major  ->  dst: [N,K] bf16 row-major
__global__ __launch_bounds__(256) void castT_f32_bf16(
    const float* __restrict__ src, unsigned short* __restrict__ dst, int K, int N) {
    __shared__ float tile[32][33];
    const int tx = threadIdx.x & 31, ty = threadIdx.x >> 5;  // 32x8
    const long n = (long)blockIdx.x * 32 + tx;
    const long kbase = (long)blockIdx.y * 32;
#pragma unroll
    for (int i = 0; i < 32; i += 8)
        tile[ty + i][tx] = src[(kbase + ty + i) * N + n];
    __syncthreads();
    const long k = kbase + tx;
    const long nbase = (long)blockIdx.x * 32;
#pragma unroll
    for (int i = 0; i < 32; i += 8)
        dst[(nbase + ty + i) * K + k] = f2bf(tile[tx][ty + i]);
}

// ---------------- RMSNorm (row of D=2048) -> bf16 ----------------
__global__ __launch_bounds__(256) void rmsnorm_bf16(const float* __restrict__ x,
                                                    const float* __restrict__ g,
                                                    unsigned short* __restrict__ out) {
    __shared__ float red[8];
    const long row = blockIdx.x;
    const float* xr = x + row * Dd;
    const int c0 = threadIdx.x * 8;
    float4 a = *(const float4*)(xr + c0);
    float4 b = *(const float4*)(xr + c0 + 4);
    float ss = a.x * a.x + a.y * a.y + a.z * a.z + a.w * a.w +
               b.x * b.x + b.y * b.y + b.z * b.z + b.w * b.w;
#pragma unroll
    for (int o = 16; o; o >>= 1) ss += __shfl_xor(ss, o);
    if ((threadIdx.x & 31) == 0) red[threadIdx.x >> 5] = ss;
    __syncthreads();
    float tot = 0.f;
#pragma unroll
    for (int i = 0; i < 8; ++i) tot += red[i];
    const float scale = rsqrtf(tot / (float)Dd + 1e-6f);
    unsigned short* orow = out + row * Dd;
    const float* gv = g + c0;
    const float in[8] = {a.x, a.y, a.z, a.w, b.x, b.y, b.z, b.w};
    ushort4 o0, o1;
    o0.x = f2bf(in[0] * scale * gv[0]); o0.y = f2bf(in[1] * scale * gv[1]);
    o0.z = f2bf(in[2] * scale * gv[2]); o0.w = f2bf(in[3] * scale * gv[3]);
    o1.x = f2bf(in[4] * scale * gv[4]); o1.y = f2bf(in[5] * scale * gv[5]);
    o1.z = f2bf(in[6] * scale * gv[6]); o1.w = f2bf(in[7] * scale * gv[7]);
    *(ushort4*)(orow + c0)     = o0;
    *(ushort4*)(orow + c0 + 4) = o1;
}

// ---------------- WMMA bf16 GEMM: C[M,N] = A[M,K] @ Bt[N,K]^T --------------
// MODE 0: outF = C
// MODE 1: outF = C + extra            (residual add)
// MODE 2: outH = bf16(silu(extra)*C)  (SwiGLU epilogue; extra = gate fp32)
#define LDSTRIDE 40  // 32 + 8 halfs pad
template <int MODE>
__global__ __launch_bounds__(256) void gemm_bf16(
    const unsigned short* __restrict__ A, const unsigned short* __restrict__ Bt,
    int M, int N, int K, const float* __restrict__ extra,
    float* __restrict__ outF, unsigned short* __restrict__ outH) {
    __shared__ unsigned short Asl[128 * LDSTRIDE];
    __shared__ unsigned short Bsl[128 * LDSTRIDE];  // [n][k]
    const int tid = threadIdx.x;
    const int lane = tid & 31, wid = tid >> 5;
    const int wm = wid & 3, wn = wid >> 2;
    const int l16 = lane & 15, lh = lane >> 4;
    const long m0 = (long)blockIdx.y * 128, n0 = (long)blockIdx.x * 128;

    v8f acc[2][4];
#pragma unroll
    for (int i = 0; i < 2; ++i)
#pragma unroll
        for (int j = 0; j < 4; ++j) acc[i][j] = zero8();

    // both tiles staged with the same geometry: row r = tid>>1, 16-half column chunk
    const int sr = tid >> 1, sch = (tid & 1) << 4;
    const unsigned short* aptr = A + (m0 + sr) * K + sch;
    const unsigned short* bptr = Bt + (n0 + sr) * K + sch;

    uint4 ag0, ag1, bg0, bg1;
    {
        const uint4* as = (const uint4*)aptr;
        const uint4* bs = (const uint4*)bptr;
        ag0 = as[0]; ag1 = as[1];
        bg0 = bs[0]; bg1 = bs[1];
    }

    for (int k0 = 0; k0 < K; k0 += 32) {
        *(uint4*)(Asl + sr * LDSTRIDE + sch)     = ag0;
        *(uint4*)(Asl + sr * LDSTRIDE + sch + 8) = ag1;
        *(uint4*)(Bsl + sr * LDSTRIDE + sch)     = bg0;
        *(uint4*)(Bsl + sr * LDSTRIDE + sch + 8) = bg1;
        __syncthreads();
        // prefetch next tile while this one computes
        if (k0 + 32 < K) {
            const uint4* as = (const uint4*)(aptr + k0 + 32);
            const uint4* bs = (const uint4*)(bptr + k0 + 32);
            ag0 = as[0]; ag1 = as[1];
            bg0 = bs[0]; bg1 = bs[1];
        }
        U16 a0, a1, bq;
        {
            const unsigned short* ap = Asl + (wm * 32 + l16) * LDSTRIDE + (lh << 3);
            a0.q[0] = *(const uint4*)ap;
            a0.q[1] = *(const uint4*)(ap + 16);
            const unsigned short* ap1 = ap + 16 * LDSTRIDE;
            a1.q[0] = *(const uint4*)ap1;
            a1.q[1] = *(const uint4*)(ap1 + 16);
        }
#pragma unroll
        for (int nt = 0; nt < 4; ++nt) {
            const unsigned short* bp = Bsl + (wn * 64 + nt * 16 + l16) * LDSTRIDE + (lh << 3);
            bq.q[0] = *(const uint4*)bp;
            bq.q[1] = *(const uint4*)(bp + 16);
            acc[0][nt] = __builtin_amdgcn_wmma_f32_16x16x32_bf16(
                false, a0.v, false, bq.v, (short)0, acc[0][nt], false, false);
            acc[1][nt] = __builtin_amdgcn_wmma_f32_16x16x32_bf16(
                false, a1.v, false, bq.v, (short)0, acc[1][nt], false, false);
        }
        __syncthreads();
    }
    // epilogue (branch-free; MODE is a template param)
#pragma unroll
    for (int mt = 0; mt < 2; ++mt)
#pragma unroll
        for (int nt = 0; nt < 4; ++nt) {
            const long col = n0 + wn * 64 + nt * 16 + l16;
            const long rbase = m0 + wm * 32 + mt * 16 + (lh << 3);
            long idx = rbase * N + col;
#pragma unroll
            for (int r = 0; r < 8; ++r, idx += N) {
                const float v = acc[mt][nt][r];
                if (MODE == 0) {
                    outF[idx] = v;
                } else if (MODE == 1) {
                    outF[idx] = v + extra[idx];
                } else {
                    const float gt = extra[idx];
                    const float s = gt / (1.f + __expf(-gt));
                    outH[idx] = f2bf(s * v);
                }
            }
        }
}

// ---------------- RoPE + relayout [B,S,NH,HD] -> [B,NH,S,HD] bf16 ----------
__global__ void rope_qk(const float* __restrict__ in, unsigned short* __restrict__ out,
                        int NH) {
    long i = (long)blockIdx.x * blockDim.x + threadIdx.x;
    const long total = (long)Bb * Ss * NH * (HDd / 2);
    if (i >= total) return;
    const int d2 = (int)(i % (HDd / 2)); long t = i / (HDd / 2);
    const int hh = (int)(t % NH); t /= NH;
    const int s = (int)(t % Ss); const int b = (int)(t / Ss);
    const float inv = __powf(10000.f, -(float)(2 * d2) / (float)HDd);
    float sn, cs;
    __sincosf((float)s * inv, &sn, &cs);
    const long ibase = (((long)b * Ss + s) * NH + hh) * HDd;
    const float x1 = in[ibase + d2], x2 = in[ibase + d2 + HDd / 2];
    const long obase = (((long)b * NH + hh) * Ss + s) * HDd;
    out[obase + d2]           = f2bf(x1 * cs - x2 * sn);
    out[obase + d2 + HDd / 2] = f2bf(x2 * cs + x1 * sn);
}

// V: [B,S,HK,HD] fp32 -> V^T [B,HK,HD,S] bf16 (writes coalesced over s)
__global__ void vcastT_kernel(const float* __restrict__ in, unsigned short* __restrict__ out) {
    long i = (long)blockIdx.x * blockDim.x + threadIdx.x;
    const long total = (long)Bb * HKk * HDd * Ss;
    if (i >= total) return;
    const int s = (int)(i % Ss); long t = i / Ss;
    const int d = (int)(t % HDd); t /= HDd;
    const int hh = (int)(t % HKk); const int b = (int)(t / HKk);
    out[i] = f2bf(in[(((long)b * Ss + s) * HKk + hh) * HDd + d]);
}

// ---------------- Flash attention (window 1024 + 4 sinks), WMMA ------------
// Q:[B,H,S,HD]  K:[B,HK,S,HD]  VT:[B,HK,HD,S] bf16 ; out:[B,S,H*HD] bf16
// One wave per 16-query tile; fragments read directly from global (KV is L2-resident).
__global__ __launch_bounds__(256) void attn_kernel(
    const unsigned short* __restrict__ Qp, const unsigned short* __restrict__ Kp,
    const unsigned short* __restrict__ VTp, unsigned short* __restrict__ Ob) {
    const int lane = threadIdx.x & 31, w = threadIdx.x >> 5;
    const int l16 = lane & 15, lh = lane >> 4;
    const int q0 = (blockIdx.x * 8 + w) * 16;
    const int h = blockIdx.y, b = blockIdx.z;
    const int kvh = h / (Hh / HKk);

    // preload Q B-fragments (Q^T operand), 4 chunks of K=32 over HD=128
    U16 qfr[4];
    const unsigned short* qrow = Qp + (((long)b * Hh + h) * Ss + q0 + l16) * HDd;
#pragma unroll
    for (int c = 0; c < 4; ++c) {
        const unsigned short* p = qrow + c * 32 + (lh << 3);
        qfr[c].q[0] = *(const uint4*)p;
        qfr[c].q[1] = *(const uint4*)(p + 16);
    }

    v8f oacc[8];
#pragma unroll
    for (int i = 0; i < 8; ++i) oacc[i] = zero8();
    float mprev = -1e30f, lsum = 0.f;

    int lo = q0 - (WINw - 1); if (lo < 0) lo = 0;
    const int lot = lo & ~31;
    const unsigned short* Kb = Kp + ((long)b * HKk + kvh) * Ss * HDd;
    const unsigned short* Vb = VTp + ((long)b * HKk + kvh) * (long)HDd * Ss;
    const int qcol = q0 + l16;

    for (int pass = 0; pass < 2; ++pass) {
        if (pass && lot == 0) break;  // sink tile already covered
        const int kstart = pass ? 0 : lot;
        const int kend = pass ? 32 : q0 + 16;
        for (int kt = kstart; kt < kend; kt += 32) {
            // S^T = K x Q^T  (two 16-key row tiles), K fragments direct from global
            v8f st[2] = {zero8(), zero8()};
#pragma unroll
            for (int ky = 0; ky < 2; ++ky) {
                int row = kt + ky * 16 + l16;
                if (row >= Ss) row = Ss - 1;  // masked anyway
                const unsigned short* krow = Kb + (long)row * HDd + (lh << 3);
#pragma unroll
                for (int c = 0; c < 4; ++c) {
                    U16 af;
                    af.q[0] = *(const uint4*)(krow + c * 32);
                    af.q[1] = *(const uint4*)(krow + c * 32 + 16);
                    st[ky] = __builtin_amdgcn_wmma_f32_16x16x32_bf16(
                        false, af.v, false, qfr[c].v, (short)0, st[ky], false, false);
                }
            }
            // mask + online softmax (stats per lane: col = query)
            const float sc = 0.088388347648318447f;  // 1/sqrt(128)
            float p[2][8];
            float tmax = -1e30f;
#pragma unroll
            for (int ky = 0; ky < 2; ++ky)
#pragma unroll
                for (int r = 0; r < 8; ++r) {
                    const int key = kt + ky * 16 + (lh << 3) + r;
                    float v = st[ky][r] * sc;
                    const bool ok = (key <= qcol) &&
                                    ((qcol - key < WINw) || (key < SINKs));
                    v = ok ? v : -1e30f;
                    p[ky][r] = v;
                    tmax = fmaxf(tmax, v);
                }
            tmax = fmaxf(tmax, __shfl_xor(tmax, 16));
            const float mnew = fmaxf(mprev, tmax);
            const float alpha = __expf(mprev - mnew);
            float rsum = 0.f;
#pragma unroll
            for (int ky = 0; ky < 2; ++ky)
#pragma unroll
                for (int r = 0; r < 8; ++r) {
                    const float e = __expf(p[ky][r] - mnew);
                    p[ky][r] = e;
                    rsum += e;
                }
            rsum += __shfl_xor(rsum, 16);
            lsum = lsum * alpha + rsum;
            mprev = mnew;
            // rescale O by per-row alpha (rows of O = queries)
#pragma unroll
            for (int r = 0; r < 8; ++r) {
                const float ar = __shfl(alpha, (lh << 3) + r);
#pragma unroll
                for (int dt = 0; dt < 8; ++dt) oacc[dt][r] *= ar;
            }
            // P fragment already in A-operand layout: e0-7 -> p[0], e8-15 -> p[1]
            U16 pf;
#pragma unroll
            for (int e = 0; e < 8; ++e) {
                pf.us[e]     = f2bf(p[0][e]);
                pf.us[e + 8] = f2bf(p[1][e]);
            }
            // O += P x V : B fragments contiguous from V^T (d-row, key-contiguous)
#pragma unroll
            for (int dt = 0; dt < 8; ++dt) {
                U16 bf;
                const unsigned short* bp = Vb + (long)(dt * 16 + l16) * Ss + kt + (lh << 3);
                bf.q[0] = *(const uint4*)bp;
                bf.q[1] = *(const uint4*)(bp + 16);
                oacc[dt] = __builtin_amdgcn_wmma_f32_16x16x32_bf16(
                    false, pf.v, false, bf.v, (short)0, oacc[dt], false, false);
            }
        }
    }
    // finalize: divide by lsum (per row) and store coalesced
    const float linv = 1.f / lsum;
#pragma unroll
    for (int r = 0; r < 8; ++r) {
        const float lr = __shfl(linv, (lh << 3) + r);
        const int qrow = q0 + (lh << 3) + r;
        unsigned short* orow = Ob + ((long)b * Ss + qrow) * (Hh * HDd) + h * HDd;
#pragma unroll
        for (int dt = 0; dt < 8; ++dt)
            orow[dt * 16 + l16] = f2bf(oacc[dt][r] * lr);
    }
}

// ---------------- host-side orchestration ----------------
static inline void* wsalloc(char*& p, size_t bytes) {
    void* r = p;
    p += (bytes + 255) & ~(size_t)255;
    return r;
}

extern "C" void kernel_launch(void* const* d_in, const int* in_sizes, int n_in,
                              void* d_out, int out_size, void* d_ws, size_t ws_size,
                              hipStream_t stream) {
    (void)in_sizes; (void)n_in; (void)out_size; (void)ws_size;
    const float* x  = (const float*)d_in[0];
    const float* g1 = (const float*)d_in[1];
    const float* g2 = (const float*)d_in[2];
    const float* wq = (const float*)d_in[3];
    const float* wk = (const float*)d_in[4];
    const float* wv = (const float*)d_in[5];
    const float* wo = (const float*)d_in[6];
    const float* wg = (const float*)d_in[7];
    const float* wu = (const float*)d_in[8];
    const float* wd = (const float*)d_in[9];
    float* out = (float*)d_out;

    const long TOK = (long)Bb * Ss;  // 4096
    char* p = (char*)d_ws;
    unsigned short* h_bf  = (unsigned short*)wsalloc(p, TOK * Dd * 2);
    unsigned short* wq_bf = (unsigned short*)wsalloc(p, (long)Dd * Hh * HDd * 2);   // [N,K]
    unsigned short* wk_bf = (unsigned short*)wsalloc(p, (long)Dd * HKk * HDd * 2);
    unsigned short* wv_bf = (unsigned short*)wsalloc(p, (long)Dd * HKk * HDd * 2);
    unsigned short* wo_bf = (unsigned short*)wsalloc(p, (long)Hh * HDd * Dd * 2);
    unsigned short* wg_bf = (unsigned short*)wsalloc(p, (long)Dd * HIDd * 2);
    unsigned short* wu_bf = (unsigned short*)wsalloc(p, (long)Dd * HIDd * 2);
    unsigned short* wd_bf = (unsigned short*)wsalloc(p, (long)HIDd * Dd * 2);
    float* qf             = (float*)wsalloc(p, TOK * Hh * HDd * 4);
    float* kf             = (float*)wsalloc(p, TOK * HKk * HDd * 4);
    float* vf             = (float*)wsalloc(p, TOK * HKk * HDd * 4);
    unsigned short* q_bf  = (unsigned short*)wsalloc(p, TOK * Hh * HDd * 2);
    unsigned short* k_bf  = (unsigned short*)wsalloc(p, TOK * HKk * HDd * 2);
    const size_t vt_bytes = (size_t)TOK * HKk * HDd * 2;
    unsigned short* vT_bf = (unsigned short*)wsalloc(p, vt_bytes + 256);  // +pad for tail overread
    unsigned short* a_bf  = (unsigned short*)wsalloc(p, TOK * Hh * HDd * 2);
    float* x1             = (float*)wsalloc(p, TOK * Dd * 4);
    unsigned short* h2_bf = (unsigned short*)wsalloc(p, TOK * Dd * 2);
    float* gatef          = (float*)wsalloc(p, TOK * HIDd * 4);
    unsigned short* actbf = (unsigned short*)wsalloc(p, TOK * HIDd * 2);

    // zero the V^T tail pad (deterministic; avoids Inf/NaN garbage in masked overreads)
    hipMemsetAsync((char*)vT_bf + vt_bytes, 0, 256, stream);

    // transpose-cast all weights: [K,N] fp32 -> [N,K] bf16
    auto castT = [&](const float* s, unsigned short* d, int K, int N) {
        castT_f32_bf16<<<dim3(N / 32, K / 32), dim3(256), 0, stream>>>(s, d, K, N);
    };
    castT(wq, wq_bf, Dd, Hh * HDd);
    castT(wk, wk_bf, Dd, HKk * HDd);
    castT(wv, wv_bf, Dd, HKk * HDd);
    castT(wo, wo_bf, Hh * HDd, Dd);
    castT(wg, wg_bf, Dd, HIDd);
    castT(wu, wu_bf, Dd, HIDd);
    castT(wd, wd_bf, HIDd, Dd);

    rmsnorm_bf16<<<dim3((unsigned)TOK), dim3(256), 0, stream>>>(x, g1, h_bf);

    auto gd = [&](int M, int N) { return dim3(N / 128, M / 128); };
    gemm_bf16<0><<<gd(TOK, Hh * HDd), dim3(256), 0, stream>>>(
        h_bf, wq_bf, (int)TOK, Hh * HDd, Dd, nullptr, qf, nullptr);
    gemm_bf16<0><<<gd(TOK, HKk * HDd), dim3(256), 0, stream>>>(
        h_bf, wk_bf, (int)TOK, HKk * HDd, Dd, nullptr, kf, nullptr);
    gemm_bf16<0><<<gd(TOK, HKk * HDd), dim3(256), 0, stream>>>(
        h_bf, wv_bf, (int)TOK, HKk * HDd, Dd, nullptr, vf, nullptr);

    {
        long tq = (long)Bb * Ss * Hh * (HDd / 2);
        rope_qk<<<dim3((unsigned)((tq + 255) / 256)), dim3(256), 0, stream>>>(qf, q_bf, Hh);
        long tk = (long)Bb * Ss * HKk * (HDd / 2);
        rope_qk<<<dim3((unsigned)((tk + 255) / 256)), dim3(256), 0, stream>>>(kf, k_bf, HKk);
        long tv = (long)Bb * HKk * HDd * Ss;
        vcastT_kernel<<<dim3((unsigned)((tv + 255) / 256)), dim3(256), 0, stream>>>(vf, vT_bf);
    }

    attn_kernel<<<dim3(Ss / 128, Hh, Bb), dim3(256), 0, stream>>>(q_bf, k_bf, vT_bf, a_bf);

    gemm_bf16<1><<<gd(TOK, Dd), dim3(256), 0, stream>>>(
        a_bf, wo_bf, (int)TOK, Dd, Hh * HDd, x, x1, nullptr);
    rmsnorm_bf16<<<dim3((unsigned)TOK), dim3(256), 0, stream>>>(x1, g2, h2_bf);
    gemm_bf16<0><<<gd(TOK, HIDd), dim3(256), 0, stream>>>(
        h2_bf, wg_bf, (int)TOK, HIDd, Dd, nullptr, gatef, nullptr);
    gemm_bf16<2><<<gd(TOK, HIDd), dim3(256), 0, stream>>>(
        h2_bf, wu_bf, (int)TOK, HIDd, Dd, gatef, nullptr, actbf);
    gemm_bf16<1><<<gd(TOK, Dd), dim3(256), 0, stream>>>(
        actbf, wd_bf, (int)TOK, Dd, HIDd, x1, out, nullptr);
}
